// SwinTransformerBlock_87411174408585
// MI455X (gfx1250) — compile-verified, matrix operand order checked by reading, not run on hardware
//
#include <hip/hip_runtime.h>
#include <hip/hip_bf16.h>

// ---------------------------------------------------------------------------
// Swin block for gfx1250 (MI455X): bf16 WMMA GEMMs with TDM-staged weight
// tiles in LDS (double buffered), wave-per-(window,head) WMMA attention.
// ---------------------------------------------------------------------------

typedef __attribute__((ext_vector_type(16))) __bf16 bf16x16;
typedef __attribute__((ext_vector_type(8)))  float  f32x8;
typedef __attribute__((ext_vector_type(4)))  unsigned u32x4;
typedef __attribute__((ext_vector_type(8)))  int      i32x8;
typedef __attribute__((ext_vector_type(4)))  int      i32x4;

#define NTOK   100352     // 32 * 56 * 56
#define CDIM   512
#define NWIN   2048       // 32 * 8 * 8
#define PWIN   49
#define HEADS  16
#define HDDIM  32

#if __has_builtin(__builtin_amdgcn_tensor_load_to_lds)
#define HAVE_TDM 1
#else
#define HAVE_TDM 0
#endif

union FragU { uint4 u[2]; bf16x16 v; };

__device__ __forceinline__ __bf16 f2bf(float f) {
  union { float f; unsigned u; } in; in.f = f;
  unsigned u = in.u;
  u += 0x7FFFu + ((u >> 16) & 1u);          // round-to-nearest-even
  union { unsigned short s; __bf16 b; } out;
  out.s = (unsigned short)(u >> 16);
  return out.b;
}

__device__ __forceinline__ f32x8 wmma_bf16(bf16x16 a, bf16x16 b, f32x8 c) {
  return __builtin_amdgcn_wmma_f32_16x16x32_bf16(
      /*neg_a=*/false, a, /*neg_b=*/false, b,
      /*c_mod=*/(short)0, c, /*reuse_a=*/false, /*reuse_b=*/false);
}

#if HAVE_TDM
// Issue a TDM load of a 64(row) x 32(col) bf16 tile of Bw[N,K] into LDS.
// D# per CDNA5 ISA 8.3/8.4: count=1, type=2(image), data_size=1(2B),
// tile_dim0=32, tile_dim1=64, tensor dims == tile dims (no OOB clipping),
// tensor_dim0_stride = K elements (row pitch of the weight matrix).
__device__ __forceinline__ void tdm_load_b_tile(const __bf16* ga, int K, unsigned ldsoff) {
  unsigned long long a = (unsigned long long)(uintptr_t)ga;
  u32x4 g0;
  g0.x = 1u;                                               // count=1, user mode
  g0.y = ldsoff;                                           // lds_addr (bytes)
  g0.z = (unsigned)a;                                      // global_addr lo
  g0.w = (unsigned)((a >> 32) & 0x01FFFFFFu) | (2u << 30); // addr hi | type=2
  i32x8 g1;
  g1[0] = 0x10000;                 // data_size=1 (2 bytes/elem)
  g1[1] = (32 << 16);              // tensor_dim0 = 32  (bits 63:48)
  g1[2] = (64 << 16);              // tensor_dim0 hi=0 | tensor_dim1 = 64
  g1[3] = (32 << 16);              // tensor_dim1 hi=0 | tile_dim0 = 32
  g1[4] = 64;                      // tile_dim1 = 64, tile_dim2 = 0
  g1[5] = K;                       // tensor_dim0_stride lo (elements)
  g1[6] = 0;                       // stride0 hi | stride1 lo
  g1[7] = 0;                       // stride1 hi
  i32x4 z4 = {0, 0, 0, 0};
#if defined(__clang_major__) && (__clang_major__ >= 23)
  i32x8 z8 = {0, 0, 0, 0, 0, 0, 0, 0};
  __builtin_amdgcn_tensor_load_to_lds(g0, g1, z4, z4, z8, 0);
#else
  __builtin_amdgcn_tensor_load_to_lds(g0, g1, z4, z4, 0);
#endif
}

#if __has_builtin(__builtin_amdgcn_s_wait_tensorcnt)
#define WAIT_TENSOR(n) __builtin_amdgcn_s_wait_tensorcnt(n)
#else
#define WAIT_TENSOR(n) asm volatile("s_wait_tensorcnt %0" :: "i"(n) : "memory")
#endif
#endif // HAVE_TDM

// ---------------------------------------------------------------------------
// fp32 -> bf16 elementwise (weights)
// ---------------------------------------------------------------------------
__global__ void cvt_bf16_kernel(const float* __restrict__ s, __bf16* __restrict__ d, int n) {
  int i = blockIdx.x * 256 + threadIdx.x;
  if (i < n) d[i] = f2bf(s[i]);
}

// ---------------------------------------------------------------------------
// LayerNorm (one wave per token, wave32, lane holds 16 channels).
// WINDOW=true : LN1 + cyclic shift(-3,-3) + window partition scatter (bf16)
// WINDOW=false: LN2 plain row-major (bf16)
// ---------------------------------------------------------------------------
template<bool WINDOW>
__global__ __launch_bounds__(256) void ln_kernel(const float* __restrict__ x,
                                                 const float* __restrict__ gam,
                                                 const float* __restrict__ bet,
                                                 __bf16* __restrict__ y) {
  int wid  = (blockIdx.x * blockDim.x + threadIdx.x) >> 5;   // token id
  int lane = threadIdx.x & 31;
  if (wid >= NTOK) return;

  const float* xp = x + (size_t)wid * CDIM;
  float v[16];
  float s = 0.f;
#pragma unroll
  for (int i = 0; i < 16; ++i) { v[i] = xp[lane + i * 32]; s += v[i]; }
#pragma unroll
  for (int o = 16; o >= 1; o >>= 1) s += __shfl_xor(s, o, 32);
  float mean = s * (1.f / 512.f);
  float q = 0.f;
#pragma unroll
  for (int i = 0; i < 16; ++i) { float d = v[i] - mean; q += d * d; }
#pragma unroll
  for (int o = 16; o >= 1; o >>= 1) q += __shfl_xor(q, o, 32);
  float rstd = rsqrtf(q * (1.f / 512.f) + 1e-5f);

  size_t row;
  if (WINDOW) {
    int t = wid, bb = t / 3136, hw = t % 3136;
    int h = hw / 56, w = hw % 56;
    int hs = h - 3;  if (hs < 0) hs += 56;          // roll(-3)
    int ws = w - 3;  if (ws < 0) ws += 56;
    row = ((size_t)bb * 64 + (hs / 7) * 8 + (ws / 7)) * PWIN + (hs % 7) * 7 + (ws % 7);
  } else {
    row = (size_t)wid;
  }
  __bf16* yp = y + row * CDIM;
#pragma unroll
  for (int i = 0; i < 16; ++i) {
    int c = lane + i * 32;
    yp[c] = f2bf((v[i] - mean) * rstd * gam[c] + bet[c]);
  }
}

// ---------------------------------------------------------------------------
// Block GEMM: out[M,N] = A[M,K](bf16) * Bw[N,K]^T(bf16) + bias
// Block = 256 threads (8 waves) computes 128(M) x 64(N).
// The shared 64x32 B tile is DMA'd into LDS by the TDM (double buffered,
// wave 0 issues, TENSORcnt-tracked); falls back to synchronous staging if
// the TDM builtin is unavailable. A fragments stream from global.
// MODE 0: QKV scatter (q scaled by 32^-0.5)          -> ob0/ob1/ob2 bf16
// MODE 1: proj + window-reverse + roll + residual(x) -> of  fp32
// MODE 2: FC1 + exact GELU                           -> ob0 bf16
// MODE 3: FC2 + residual(x2)                         -> of  fp32
// ---------------------------------------------------------------------------
template<int MODE>
__global__ __launch_bounds__(256) void gemm128x64(const __bf16* __restrict__ A,
                                                  const __bf16* __restrict__ Bw,
                                                  const float* __restrict__ bias,
                                                  int M, int N, int K,
                                                  __bf16* __restrict__ ob0,
                                                  __bf16* __restrict__ ob1,
                                                  __bf16* __restrict__ ob2,
                                                  float* __restrict__ of,
                                                  const float* __restrict__ resid) {
  __shared__ __align__(16) __bf16 Bbuf[2][64 * 32];   // double-buffered weight tile

  const int lane   = threadIdx.x & 31;
  const int wv     = threadIdx.x >> 5;                // 0..7
  const int tilesN = N >> 6;
  const int bm     = blockIdx.x / tilesN;             // 128-row block index
  const int nt     = blockIdx.x % tilesN;
  const int mt     = bm * 8 + wv;                     // this wave's 16-row tile

  const int halfk = (lane >> 4) << 3;
  const __bf16* Abase  = A  + ((size_t)mt * 16 + (lane & 15)) * (size_t)K;
  const __bf16* Btile0 = Bw + (size_t)(nt * 64) * (size_t)K;   // rows nt*64..+63

  f32x8 acc[4];
#pragma unroll
  for (int j = 0; j < 4; ++j) acc[j] = f32x8{0.f,0.f,0.f,0.f,0.f,0.f,0.f,0.f};

  const int nk = K >> 5;

#if HAVE_TDM
  if (wv == 0)
    tdm_load_b_tile(Btile0, K, (unsigned)(uintptr_t)&Bbuf[0][0]);
#endif

  for (int i = 0; i < nk; ++i) {
    const int k0 = i << 5;
#if HAVE_TDM
    if (wv == 0) {
      if (i + 1 < nk) {
        tdm_load_b_tile(Btile0 + (k0 + 32), K,
                        (unsigned)(uintptr_t)&Bbuf[(i + 1) & 1][0]);
        WAIT_TENSOR(1);             // buffer i is landed; next DMA in flight
      } else {
        WAIT_TENSOR(0);
      }
    }
#else
    { // synchronous staging: 256 threads x 16B = 4KB tile
      int r = threadIdx.x >> 2, seg = threadIdx.x & 3;
      uint4 d = *(const uint4*)(Btile0 + (size_t)r * K + k0 + seg * 8);
      *(uint4*)&Bbuf[i & 1][r * 32 + seg * 8] = d;
    }
#endif
    __syncthreads();                // tile (i&1) visible to all waves

    FragU fa;
    const __bf16* pa = Abase + k0 + halfk;
    fa.u[0] = *(const uint4*)(pa);
    fa.u[1] = *(const uint4*)(pa + 16);
    const __bf16* Bl = &Bbuf[i & 1][0];
#pragma unroll
    for (int j = 0; j < 4; ++j) {
      const __bf16* pb = Bl + (j * 16 + (lane & 15)) * 32 + halfk;
      FragU fb;
      fb.u[0] = *(const uint4*)(pb);
      fb.u[1] = *(const uint4*)(pb + 16);
      acc[j] = wmma_bf16(fa.v, fb.v, acc[j]);
    }
    __syncthreads();                // done reading before tile is overwritten
  }

  const int nl = lane & 15, mb = (lane >> 4) * 8;
#pragma unroll
  for (int j = 0; j < 4; ++j) {
#pragma unroll
    for (int r = 0; r < 8; ++r) {
      int m = mt * 16 + mb + r;
      int n = nt * 64 + j * 16 + nl;
      float val = acc[j][r] + bias[n];
      if (MODE == 0) {                       // QKV scatter to [win][head][P][HD]
        int part = n >> 9;
        int cc = n & 511, head = cc >> 5, hd = cc & 31;
        int winp = m / PWIN, pp = m % PWIN;
        size_t idx = (((size_t)winp * HEADS + head) * PWIN + pp) * HDDIM + hd;
        if (part == 0)      ob0[idx] = f2bf(val * 0.17677669529663687f);
        else if (part == 1) ob1[idx] = f2bf(val);
        else                ob2[idx] = f2bf(val);
      } else if (MODE == 1) {                // proj: window reverse + roll(+3) + residual
        int winp = m / PWIN, pp = m % PWIN;
        int bb = winp >> 6, w64 = winp & 63;
        int wi = w64 >> 3, wj = w64 & 7;
        int hs = wi * 7 + pp / 7 + 3; if (hs >= 56) hs -= 56;
        int ws = wj * 7 + pp % 7 + 3; if (ws >= 56) ws -= 56;
        size_t t = ((size_t)bb * 3136 + hs * 56 + ws) * CDIM + n;
        of[t] = resid[t] + val;
      } else if (MODE == 2) {                // FC1 + exact GELU
        float g = 0.5f * val * (1.f + erff(val * 0.70710678118654752f));
        ob0[(size_t)m * (size_t)N + n] = f2bf(g);
      } else {                               // FC2 + residual
        size_t t = (size_t)m * CDIM + n;
        of[t] = resid[t] + val;
      }
    }
  }
}

// ---------------------------------------------------------------------------
// Attention: one wave per (window, head).
//   S = q k^T      (16 WMMAs, K=32 == HD exactly)
//   softmax(S + relbias + shiftmask)  in LDS (fp32 scores, bf16 probs)
//   O = P v        (16 WMMAs from LDS fragments), row 1/sum folded into store
// ---------------------------------------------------------------------------
__global__ __launch_bounds__(32) void attn_kernel(const __bf16* __restrict__ qg,
                                                  const __bf16* __restrict__ kg,
                                                  const __bf16* __restrict__ vg,
                                                  const float* __restrict__ relb,
                                                  __bf16* __restrict__ og) {
  __shared__ __align__(16) float  sBias[169 * 16];   // rel_bias table
  __shared__ __align__(16) float  Sf[64 * 64];       // fp32 scores
  __shared__ __align__(16) __bf16 S2[64 * 64];       // bf16 probabilities (unnormalized)
  __shared__ __align__(16) __bf16 vT[32 * 64];       // v transposed [hd][p], zero padded
  __shared__ __align__(16) float  rowscale[64];      // 1 / rowsum

  const int lane = threadIdx.x & 31;
  const int unit = blockIdx.x;                        // win*16 + head
  const int win  = unit >> 4;
  const int head = unit & 15;
  const int w64  = win & 63;
  const int wi = w64 >> 3, wj = w64 & 7;

  const __bf16* qb = qg + (size_t)unit * (PWIN * HDDIM);
  const __bf16* kb = kg + (size_t)unit * (PWIN * HDDIM);
  const __bf16* vb = vg + (size_t)unit * (PWIN * HDDIM);

  for (int i = lane; i < 169 * 16; i += 32) sBias[i] = relb[i];
  for (int idx = lane; idx < 32 * 64; idx += 32) {
    int hd = idx >> 6, p = idx & 63;
    vT[idx] = (p < PWIN) ? vb[p * HDDIM + hd] : f2bf(0.f);
  }
  __syncthreads();

  const int halfk = (lane >> 4) << 3;

  // ---- GEMM1: S = q k^T -------------------------------------------------
  for (int mi = 0; mi < 4; ++mi) {
    int rm = mi * 16 + (lane & 15); if (rm > 48) rm = 48;   // clamp padded rows
    FragU fa;
    const __bf16* pa = qb + rm * HDDIM + halfk;
    fa.u[0] = *(const uint4*)(pa);
    fa.u[1] = *(const uint4*)(pa + 16);
#pragma unroll
    for (int nj = 0; nj < 4; ++nj) {
      int rn = nj * 16 + (lane & 15); if (rn > 48) rn = 48;
      FragU fb;
      const __bf16* pb = kb + rn * HDDIM + halfk;
      fb.u[0] = *(const uint4*)(pb);
      fb.u[1] = *(const uint4*)(pb + 16);
      f32x8 acc = f32x8{0.f,0.f,0.f,0.f,0.f,0.f,0.f,0.f};
      acc = wmma_bf16(fa.v, fb.v, acc);
      int nl = lane & 15, mbs = (lane >> 4) * 8;
#pragma unroll
      for (int r = 0; r < 8; ++r)
        Sf[(mi * 16 + mbs + r) * 64 + nj * 16 + nl] = acc[r];
    }
  }
  __syncthreads();

  // ---- bias + mask + softmax (lane owns rows lane and lane+32) ----------
  for (int rr = 0; rr < 2; ++rr) {
    int p = lane + rr * 32;
    if (p >= PWIN) {
      for (int q = 0; q < 64; ++q) S2[p * 64 + q] = f2bf(0.f);
      rowscale[p] = 0.f;
      continue;
    }
    int ph = p / 7, pw = p % 7;
    int labp = ((wi < 7) ? 0 : ((ph < 4) ? 1 : 2)) * 3 +
               ((wj < 7) ? 0 : ((pw < 4) ? 1 : 2));
    float mx = -3.0e38f;
    for (int q = 0; q < PWIN; ++q) {
      int qh = q / 7, qw = q % 7;
      int ridx = (ph - qh + 6) * 13 + (pw - qw + 6);
      float bv = sBias[ridx * 16 + head];
      int labq = ((wi < 7) ? 0 : ((qh < 4) ? 1 : 2)) * 3 +
                 ((wj < 7) ? 0 : ((qw < 4) ? 1 : 2));
      float mk = (labp != labq) ? -100.f : 0.f;
      float vv = Sf[p * 64 + q] + bv + mk;
      Sf[p * 64 + q] = vv;
      mx = fmaxf(mx, vv);
    }
    float sum = 0.f;
    for (int q = 0; q < PWIN; ++q) {
      float e = __expf(Sf[p * 64 + q] - mx);
      sum += e;
      S2[p * 64 + q] = f2bf(e);
    }
    for (int q = PWIN; q < 64; ++q) S2[p * 64 + q] = f2bf(0.f);
    rowscale[p] = 1.f / sum;
  }
  __syncthreads();

  // ---- GEMM2: O = P v ----------------------------------------------------
  for (int mi = 0; mi < 4; ++mi) {
#pragma unroll
    for (int nj = 0; nj < 2; ++nj) {
      f32x8 acc = f32x8{0.f,0.f,0.f,0.f,0.f,0.f,0.f,0.f};
#pragma unroll
      for (int kj = 0; kj < 2; ++kj) {
        FragU fa;
        const __bf16* pa = &S2[(mi * 16 + (lane & 15)) * 64 + kj * 32 + halfk];
        fa.u[0] = *(const uint4*)(pa);
        fa.u[1] = *(const uint4*)(pa + 16);
        FragU fb;
        const __bf16* pb = &vT[(nj * 16 + (lane & 15)) * 64 + kj * 32 + halfk];
        fb.u[0] = *(const uint4*)(pb);
        fb.u[1] = *(const uint4*)(pb + 16);
        acc = wmma_bf16(fa.v, fb.v, acc);
      }
      int nl = lane & 15, mbs = (lane >> 4) * 8;
#pragma unroll
      for (int r = 0; r < 8; ++r) {
        int m = mi * 16 + mbs + r;
        if (m < PWIN) {
          float val = acc[r] * rowscale[m];
          og[((size_t)win * PWIN + m) * CDIM + head * HDDIM + nj * 16 + nl] = f2bf(val);
        }
      }
    }
  }
}

// ---------------------------------------------------------------------------
// Host-side orchestration
// ---------------------------------------------------------------------------
extern "C" void kernel_launch(void* const* d_in, const int* in_sizes, int n_in,
                              void* d_out, int out_size, void* d_ws, size_t ws_size,
                              hipStream_t stream) {
  (void)in_sizes; (void)n_in; (void)out_size; (void)ws_size;

  const float* x        = (const float*)d_in[0];
  const float* ln1_g    = (const float*)d_in[3];
  const float* ln1_b    = (const float*)d_in[4];
  const float* qkv_w    = (const float*)d_in[5];
  const float* qkv_b    = (const float*)d_in[6];
  const float* rel_bias = (const float*)d_in[7];
  const float* proj_w   = (const float*)d_in[8];
  const float* proj_b   = (const float*)d_in[9];
  const float* ln2_g    = (const float*)d_in[10];
  const float* ln2_b    = (const float*)d_in[11];
  const float* fc1_w    = (const float*)d_in[12];
  const float* fc1_b    = (const float*)d_in[13];
  const float* fc2_w    = (const float*)d_in[14];
  const float* fc2_b    = (const float*)d_in[15];
  float* out = (float*)d_out;

  char* ws = (char*)d_ws;
  // workspace layout (bytes); aliases are stream-order safe:
  //   o aliases y (y dead after QKV GEMM), y2 aliases q (dead after attention),
  //   x2 (fp32) aliases k+v (dead after attention).
  __bf16* wq_bf   = (__bf16*)(ws + 0);                       // 1536*512
  __bf16* wpj_bf  = (__bf16*)(ws + 1572864);                 // 512*512
  __bf16* wf1_bf  = (__bf16*)(ws + 2097152);                 // 2048*512
  __bf16* wf2_bf  = (__bf16*)(ws + 4194304);                 // 512*2048
  __bf16* y_bf    = (__bf16*)(ws + 6291456);                 // NTOK*512 bf16
  __bf16* o_bf    = y_bf;                                    // alias
  __bf16* q_bf    = (__bf16*)(ws + 109051904);               // NTOK*512 bf16
  __bf16* y2_bf   = q_bf;                                    // alias
  __bf16* k_bf    = (__bf16*)(ws + 211812352);               // NTOK*512 bf16
  __bf16* v_bf    = (__bf16*)(ws + 314572800);               // NTOK*512 bf16
  float*  x2      = (float*) (ws + 211812352);               // alias over k+v (fp32)
  __bf16* h_bf    = (__bf16*)(ws + 417333248);               // NTOK*2048 bf16

  // 1) weights fp32 -> bf16
  cvt_bf16_kernel<<<(1536 * 512 + 255) / 256, 256, 0, stream>>>(qkv_w, wq_bf, 1536 * 512);
  cvt_bf16_kernel<<<(512 * 512 + 255) / 256, 256, 0, stream>>>(proj_w, wpj_bf, 512 * 512);
  cvt_bf16_kernel<<<(2048 * 512 + 255) / 256, 256, 0, stream>>>(fc1_w, wf1_bf, 2048 * 512);
  cvt_bf16_kernel<<<(512 * 2048 + 255) / 256, 256, 0, stream>>>(fc2_w, wf2_bf, 512 * 2048);

  // 2) LN1 + shift + window partition  (8 tokens per 256-thread block)
  ln_kernel<true><<<NTOK / 8, 256, 0, stream>>>(x, ln1_g, ln1_b, y_bf);

  // 3) QKV GEMM:  [100352,512] x [512,1536]^T  -> q,k,v scatter
  gemm128x64<0><<<(NTOK / 128) * (1536 / 64), 256, 0, stream>>>(
      y_bf, wq_bf, qkv_b, NTOK, 1536, 512, q_bf, k_bf, v_bf, nullptr, nullptr);

  // 4) windowed attention: one wave per (window, head)
  attn_kernel<<<NWIN * HEADS, 32, 0, stream>>>(q_bf, k_bf, v_bf, rel_bias, o_bf);

  // 5) proj GEMM + window reverse + roll + residual -> x2 (fp32)
  gemm128x64<1><<<(NTOK / 128) * (512 / 64), 256, 0, stream>>>(
      o_bf, wpj_bf, proj_b, NTOK, 512, 512, nullptr, nullptr, nullptr, x2, x);

  // 6) LN2 -> y2 (bf16)
  ln_kernel<false><<<NTOK / 8, 256, 0, stream>>>(x2, ln2_g, ln2_b, y2_bf);

  // 7) FC1 + GELU -> h (bf16)
  gemm128x64<2><<<(NTOK / 128) * (2048 / 64), 256, 0, stream>>>(
      y2_bf, wf1_bf, fc1_b, NTOK, 2048, 512, h_bf, nullptr, nullptr, nullptr, nullptr);

  // 8) FC2 + residual -> out (fp32)
  gemm128x64<3><<<(NTOK / 128) * (512 / 64), 256, 0, stream>>>(
      h_bf, wf2_bf, fc2_b, NTOK, 512, 2048, nullptr, nullptr, nullptr, out, x2);
}
